// DLRM_Net_60301340835920
// MI455X (gfx1250) — matrix operand order repaired and checked by reading, not compiled, not run on hardware
//
#include <hip/hip_runtime.h>
#include <hip/hip_bf16.h>

typedef unsigned short u16;
typedef __attribute__((ext_vector_type(16))) __bf16 v16bf;
typedef __attribute__((ext_vector_type(16))) u16    v16u;
typedef __attribute__((ext_vector_type(8)))  float  v8f;

#define BATCH 32768
#define NTAB  26
#define NROWS 100000
#define EMBD  128

__device__ __forceinline__ u16 f2bf(float f) {
    unsigned u = __builtin_bit_cast(unsigned, f);
    unsigned r = (u + 0x7FFFu + ((u >> 16) & 1u)) >> 16;
    return (u16)r;
}
__device__ __forceinline__ float bf2f(u16 h) {
    unsigned u = ((unsigned)h) << 16;
    return __builtin_bit_cast(float, u);
}

// ---------------------------------------------------------------------------
// Pad + convert f32 (N x Ks, row major) -> bf16 (N x Kd, zero-padded cols)
// ---------------------------------------------------------------------------
__global__ void pad_cvt_kernel(const float* __restrict__ src, u16* __restrict__ dst,
                               int N, int Ks, int Kd) {
    size_t i = (size_t)blockIdx.x * blockDim.x + threadIdx.x;
    size_t total = (size_t)N * (size_t)Kd;
    if (i >= total) return;
    int n = (int)(i / Kd);
    int k = (int)(i % Kd);
    dst[i] = (k < Ks) ? f2bf(src[(size_t)n * Ks + k]) : (u16)0;
}

// ---------------------------------------------------------------------------
// Y = relu?(X @ W^T + b)   X: BATCH x Kpad bf16, W: N x Kpad bf16 (row major),
// Y: BATCH x N bf16.  One wave computes a 32(M) x 64(N) output tile:
// 2 A-tiles x 4 B-tiles -> 8 accumulators, 8 v_wmma_f32_16x16x32_bf16 per
// K-step (K stepped by 32).  Grid: (N/64, BATCH/32/8), block 256 (8 waves).
// ---------------------------------------------------------------------------
__global__ __launch_bounds__(256) void gemm_bf16_kernel(
    const u16* __restrict__ X, const u16* __restrict__ W,
    const float* __restrict__ bias, u16* __restrict__ Y,
    int Kpad, int N, int do_relu) {

    const int lane  = threadIdx.x & 31;
    const int wid   = threadIdx.x >> 5;
    const int mbase = (blockIdx.y * 8 + wid) * 32;  // 32 rows per wave
    const int nbase = blockIdx.x * 64;              // 64 cols per wave
    const int hi    = lane >> 4;                    // K-half selector
    const int l15   = lane & 15;                    // A: row-in-tile, B: col-in-tile

    const u16* xrow0 = X + (size_t)(mbase + l15) * Kpad;
    const u16* xrow1 = X + (size_t)(mbase + 16 + l15) * Kpad;
    const u16* wrow[4];
    #pragma unroll
    for (int nt = 0; nt < 4; ++nt)
        wrow[nt] = W + (size_t)(nbase + nt * 16 + l15) * Kpad;

    v8f acc[2][4];
    #pragma unroll
    for (int mt = 0; mt < 2; ++mt)
        #pragma unroll
        for (int nt = 0; nt < 4; ++nt)
            acc[mt][nt] = (v8f){};

    for (int k0 = 0; k0 < Kpad; k0 += 32) {
        // A (16x32 bf16 per tile): halves 0..7 = K[k0+hi*8..+8),
        // halves 8..15 = K[k0+16+hi*8..+8) (ISA 16-bit A layout).
        v16bf a[2];
        {
            v16u au0, au1;
            #pragma unroll
            for (int j = 0; j < 8; ++j) {
                au0[j]     = xrow0[k0 + hi * 8 + j];
                au0[8 + j] = xrow0[k0 + 16 + hi * 8 + j];
                au1[j]     = xrow1[k0 + hi * 8 + j];
                au1[8 + j] = xrow1[k0 + 16 + hi * 8 + j];
            }
            a[0] = __builtin_bit_cast(v16bf, au0);
            a[1] = __builtin_bit_cast(v16bf, au1);
        }
        // B (32x16 bf16 per tile): column n=l15; halves 0..7 = K[k0+hi*8..+8),
        // halves 8..15 = K[k0+16+hi*8..+8) (B[k][n] = W[n][k], W row-major N x K).
        v16bf b[4];
        #pragma unroll
        for (int nt = 0; nt < 4; ++nt) {
            v16u bu;
            #pragma unroll
            for (int j = 0; j < 8; ++j) { bu[j] = wrow[nt][k0 + hi * 8 + j]; bu[8 + j] = wrow[nt][k0 + 16 + hi * 8 + j]; }
            b[nt] = __builtin_bit_cast(v16bf, bu);
        }
        #pragma unroll
        for (int mt = 0; mt < 2; ++mt)
            #pragma unroll
            for (int nt = 0; nt < 4; ++nt)
                acc[mt][nt] = __builtin_amdgcn_wmma_f32_16x16x32_bf16(
                    false, a[mt], false, b[nt], (short)0, acc[mt][nt], false, false);
    }

    #pragma unroll
    for (int nt = 0; nt < 4; ++nt) {
        const int nOut = nbase + nt * 16 + l15;
        const float bn = bias[nOut];
        #pragma unroll
        for (int mt = 0; mt < 2; ++mt) {
            #pragma unroll
            for (int i = 0; i < 8; ++i) {
                int mOut = mbase + mt * 16 + i + hi * 8;  // C/D: M = vgpr + 8*hi
                float v = acc[mt][nt][i] + bn;
                if (do_relu) v = v > 0.f ? v : 0.f;
                Y[(size_t)mOut * N + nOut] = f2bf(v);
            }
        }
    }
}

// ---------------------------------------------------------------------------
// Embedding gather + pairwise interaction.
// One wave per sample.  T (32x128 bf16 in LDS): row0 = x, rows1..26 = emb
// gathers, rows27..31 = 0.  Z = T @ T^T via 2x2 WMMA tiles, K=128.
// R[b] (512 bf16) = [ x(128) | tril(Z,k=-1) flat (351) | zeros(33) ].
// Block 128 threads = 4 waves, LDS 32 KB.
// ---------------------------------------------------------------------------
__global__ __launch_bounds__(128) void interact_kernel(
    const u16* __restrict__ Xb, const float* __restrict__ embW,
    const long long* __restrict__ lSi, u16* __restrict__ R) {

    __shared__ u16 Tsh[4][32 * EMBD];
    const int lane = threadIdx.x & 31;
    const int wid  = threadIdx.x >> 5;
    const int b    = blockIdx.x * 4 + wid;
    u16* T = Tsh[wid];

    // row 0: bottom-MLP output x (already bf16)
    {
        const u16* src = Xb + (size_t)b * EMBD;
        #pragma unroll
        for (int j = 0; j < 4; ++j) T[lane * 4 + j] = src[lane * 4 + j];
    }
    // rows 1..26: gather fp32 embedding rows, convert to bf16
    for (int t = 0; t < NTAB; ++t) {
        long long idx = lSi[(size_t)t * BATCH + b];
        const float* src = embW + ((size_t)t * NROWS + (size_t)idx) * EMBD;
        u16* dst = T + (t + 1) * EMBD;
        #pragma unroll
        for (int j = 0; j < 4; ++j) dst[lane * 4 + j] = f2bf(src[lane * 4 + j]);
    }
    // rows 27..31: zero pad
    for (int r = NTAB + 1; r < 32; ++r) {
        #pragma unroll
        for (int j = 0; j < 4; ++j) T[r * EMBD + lane * 4 + j] = 0;
    }
    __syncthreads();

    const int hi  = lane >> 4;
    const int l15 = lane & 15;
    v8f acc00 = {}, acc01 = {}, acc10 = {}, acc11 = {};

    for (int k0 = 0; k0 < EMBD; k0 += 32) {
        v16bf a[2], bm[2];
        #pragma unroll
        for (int tm = 0; tm < 2; ++tm) {
            const u16* tr = T + (size_t)(tm * 16 + l15) * EMBD;
            v16u au;
            #pragma unroll
            for (int j = 0; j < 8; ++j) {
                au[j]     = tr[k0 + hi * 8 + j];
                au[8 + j] = tr[k0 + 16 + hi * 8 + j];
            }
            a[tm] = __builtin_bit_cast(v16bf, au);
        }
        #pragma unroll
        for (int tn = 0; tn < 2; ++tn) {
            const u16* tc = T + (size_t)(tn * 16 + l15) * EMBD;  // B[k][n]=T[n][k]
            v16u bu;
            #pragma unroll
            for (int j = 0; j < 8; ++j) { bu[j] = tc[k0 + hi * 8 + j]; bu[8 + j] = tc[k0 + 16 + hi * 8 + j]; }
            bm[tn] = __builtin_bit_cast(v16bf, bu);
        }
        acc00 = __builtin_amdgcn_wmma_f32_16x16x32_bf16(false, a[0], false, bm[0], (short)0, acc00, false, false);
        acc01 = __builtin_amdgcn_wmma_f32_16x16x32_bf16(false, a[0], false, bm[1], (short)0, acc01, false, false);
        acc10 = __builtin_amdgcn_wmma_f32_16x16x32_bf16(false, a[1], false, bm[0], (short)0, acc10, false, false);
        acc11 = __builtin_amdgcn_wmma_f32_16x16x32_bf16(false, a[1], false, bm[1], (short)0, acc11, false, false);
    }

    u16* rrow = R + (size_t)b * 512;
    // R[0..127] = x
    #pragma unroll
    for (int j = 0; j < 4; ++j) rrow[lane * 4 + j] = T[lane * 4 + j];
    // R[479..511] = 0 (two overlapping strided stores cover all 33 entries)
    rrow[479 + lane] = 0;
    rrow[480 + lane] = 0;
    // scatter strict lower triangle: flat index i*(i-1)/2 + j (tril k=-1 order)
    const v8f* accs[4] = { &acc00, &acc01, &acc10, &acc11 };
    #pragma unroll
    for (int tm = 0; tm < 2; ++tm)
        #pragma unroll
        for (int tn = 0; tn < 2; ++tn) {
            const v8f& a4 = *accs[tm * 2 + tn];
            #pragma unroll
            for (int i = 0; i < 8; ++i) {
                int Mi = tm * 16 + i + hi * 8;
                int Nj = tn * 16 + l15;
                if (Mi > Nj && Mi <= NTAB) {
                    int fidx = (Mi * (Mi - 1)) / 2 + Nj;
                    rrow[128 + fidx] = f2bf(a4[i]);
                }
            }
        }
}

// ---------------------------------------------------------------------------
// Final layer: p = sigmoid(A(1024, bf16) . w(1024, f32) + b).  One wave/row.
// ---------------------------------------------------------------------------
__global__ __launch_bounds__(256) void top2_sigmoid_kernel(
    const u16* __restrict__ A, const float* __restrict__ w,
    const float* __restrict__ bias, float* __restrict__ out) {
    const int lane = threadIdx.x & 31;
    const int wid  = threadIdx.x >> 5;
    const int b    = blockIdx.x * 8 + wid;
    const u16* arow = A + (size_t)b * 1024;
    float s = 0.f;
    for (int j = lane; j < 1024; j += 32) s += bf2f(arow[j]) * w[j];
    #pragma unroll
    for (int d = 16; d > 0; d >>= 1) s += __shfl_xor(s, d, 32);
    if (lane == 0) out[b] = 1.f / (1.f + __expf(-(s + bias[0])));
}

// ---------------------------------------------------------------------------
extern "C" void kernel_launch(void* const* d_in, const int* in_sizes, int n_in,
                              void* d_out, int out_size, void* d_ws, size_t ws_size,
                              hipStream_t stream) {
    (void)in_sizes; (void)n_in; (void)out_size; (void)ws_size;
    const float*     dense_x = (const float*)d_in[0];
    const long long* lS_i    = (const long long*)d_in[1];
    // d_in[2] = lS_o (unused)
    const float*     emb_W   = (const float*)d_in[3];
    const float *bw0 = (const float*)d_in[4],  *bb0 = (const float*)d_in[5];
    const float *bw1 = (const float*)d_in[6],  *bb1 = (const float*)d_in[7];
    const float *bw2 = (const float*)d_in[8],  *bb2 = (const float*)d_in[9];
    const float *tw0 = (const float*)d_in[10], *tb0 = (const float*)d_in[11];
    const float *tw1 = (const float*)d_in[12], *tb1 = (const float*)d_in[13];
    const float *tw2 = (const float*)d_in[14], *tb2 = (const float*)d_in[15];
    float* out = (float*)d_out;

    // ---- workspace layout (bytes, 256-aligned chunks) ----
    char* ws = (char*)d_ws;
    size_t off = 0;
    auto alloc = [&](size_t bytes) { char* p = ws + off; off = (off + bytes + 255) & ~(size_t)255; return p; };
    u16* W0P = (u16*)alloc((size_t)512 * 32 * 2);
    u16* W1P = (u16*)alloc((size_t)256 * 512 * 2);
    u16* W2P = (u16*)alloc((size_t)128 * 256 * 2);
    u16* T0P = (u16*)alloc((size_t)1024 * 512 * 2);
    u16* T1P = (u16*)alloc((size_t)1024 * 1024 * 2);
    u16* X0P = (u16*)alloc((size_t)BATCH * 32 * 2);
    u16* A1  = (u16*)alloc((size_t)BATCH * 512 * 2);
    u16* A2  = (u16*)alloc((size_t)BATCH * 256 * 2);
    u16* XB  = (u16*)alloc((size_t)BATCH * 128 * 2);
    u16* RB  = (u16*)alloc((size_t)BATCH * 512 * 2);
    u16* Z1  = (u16*)alloc((size_t)BATCH * 1024 * 2);
    u16* Z2  = (u16*)alloc((size_t)BATCH * 1024 * 2);

    auto padN = [](size_t total) { return (unsigned)((total + 255) / 256); };

    // ---- weight / input conversion (f32 -> padded bf16) ----
    pad_cvt_kernel<<<padN((size_t)512 * 32), 256, 0, stream>>>(bw0, W0P, 512, 13, 32);
    pad_cvt_kernel<<<padN((size_t)256 * 512), 256, 0, stream>>>(bw1, W1P, 256, 512, 512);
    pad_cvt_kernel<<<padN((size_t)128 * 256), 256, 0, stream>>>(bw2, W2P, 128, 256, 256);
    pad_cvt_kernel<<<padN((size_t)1024 * 512), 256, 0, stream>>>(tw0, T0P, 1024, 479, 512);
    pad_cvt_kernel<<<padN((size_t)1024 * 1024), 256, 0, stream>>>(tw1, T1P, 1024, 1024, 1024);
    pad_cvt_kernel<<<padN((size_t)BATCH * 32), 256, 0, stream>>>(dense_x, X0P, BATCH, 13, 32);

    // ---- bottom MLP ----  grid: (N/64, BATCH/32/8)
    gemm_bf16_kernel<<<dim3(512 / 64, BATCH / 32 / 8), 256, 0, stream>>>(X0P, W0P, bb0, A1, 32, 512, 1);
    gemm_bf16_kernel<<<dim3(256 / 64, BATCH / 32 / 8), 256, 0, stream>>>(A1, W1P, bb1, A2, 512, 256, 1);
    gemm_bf16_kernel<<<dim3(128 / 64, BATCH / 32 / 8), 256, 0, stream>>>(A2, W2P, bb2, XB, 256, 128, 1);

    // ---- embedding gather + pairwise interaction ----
    interact_kernel<<<BATCH / 4, 128, 0, stream>>>(XB, emb_W, lS_i, RB);

    // ---- top MLP ----
    gemm_bf16_kernel<<<dim3(1024 / 64, BATCH / 32 / 8), 256, 0, stream>>>(RB, T0P, tb0, Z1, 512, 1024, 1);
    gemm_bf16_kernel<<<dim3(1024 / 64, BATCH / 32 / 8), 256, 0, stream>>>(Z1, T1P, tb1, Z2, 1024, 1024, 1);
    top2_sigmoid_kernel<<<BATCH / 8, 256, 0, stream>>>(Z2, tw2, tb2, out);
}
